// MultiConv1d_19679540150835
// MI455X (gfx1250) — matrix operand order repaired and checked by reading, not compiled, not run on hardware
//
#include <hip/hip_runtime.h>

// MultiConv1d (grouped conv1d, one group per instance) for MI455X / gfx1250.
// B=32 T=128 N=200 FI=32 FO=64 K=3, stride=1 pad=1 dil=1.
// Per-(b,n) workgroup GEMM [T,96]x[96,64]; reduction 96 = 3 k-major chunks of
// 32 so each A chunk is a time-shifted window of the staged x slab (no im2col).
// bf16 WMMA with hi/lo split (3-term emulation, ~f32 accuracy).
// All f32->bf16 splitting + fragment layout is done ONCE at staging time with
// pure bit-op index decode; the hot loop is pure ds_load_b128 -> v_wmma.

typedef __attribute__((ext_vector_type(16))) __bf16 v16bf;
typedef __attribute__((ext_vector_type(8)))  float  v8f;

#define Bb   32
#define Tt   128
#define Nn   200
#define FIc  32
#define FOc  64
#define KK   3
#define XROWS 130      // t = -1 .. 128 (zero halo rows at both ends)
#define XPITCH 40      // bf16 per row: 32 data + 8 pad (80B, 16B aligned)
#define BPITCH 40      // bf16 per lane block: 16 hi + 16 lo + 8 pad (80B)

__device__ __forceinline__ unsigned f32bits(float f) {
  return __builtin_bit_cast(unsigned, f);
}
__device__ __forceinline__ float bf_hi_f32(float f) {
  return __builtin_bit_cast(float, f32bits(f) & 0xFFFF0000u);
}
// pack two f32 (truncated to bf16) into one dword: e0 -> low half, e1 -> high
__device__ __forceinline__ unsigned bf_pack2(float f0, float f1) {
  return (f32bits(f0) >> 16) | (f32bits(f1) & 0xFFFF0000u);
}

union FragB { v16bf v; uint4 q[2]; };

__global__ __launch_bounds__(256)
void multiconv1d_wmma_bf16x3(const float* __restrict__ x,
                             const float* __restrict__ w,
                             const float* __restrict__ bias,
                             float* __restrict__ out)
{
  // x slab pre-split to bf16 hi/lo, row r == t+1 (rows 0 and 129 are zero).
  __shared__ __align__(16) __bf16 lds_xh[XROWS][XPITCH];
  __shared__ __align__(16) __bf16 lds_xl[XROWS][XPITCH];
  // Weight B-fragments in exact WMMA lane layout:
  // [(c*4+nt)][lane][0:15]=hi, [16:31]=lo  (lane stride BPITCH bf16 = 80B)
  __shared__ __align__(16) __bf16 lds_b[KK * 4 * 32 * BPITCH];

  const int b   = blockIdx.x;   // fast dim: same-n blocks adjacent (weight L2 reuse)
  const int n   = blockIdx.y;
  const int tid = threadIdx.x;

  // ---- stage x[b,:,n,0:32]: float4 loads, split to bf16 hi/lo rows ----
  {
    const float* xbase = x + ((size_t)(b * Tt) * Nn + n) * FIc;
    const int q  = tid & 7;      // 16B quad within a 32-float row
    const int r0 = tid >> 3;     // 0..31
#pragma unroll
    for (int p = 0; p < 4; ++p) {
      const int row = r0 + p * 32;                 // 0..127
      const float4 v = *(const float4*)(xbase + (size_t)row * (Nn * FIc) + q * 4);
      uint2 hp, lp;
      hp.x = bf_pack2(v.x, v.y);
      hp.y = bf_pack2(v.z, v.w);
      lp.x = bf_pack2(v.x - bf_hi_f32(v.x), v.y - bf_hi_f32(v.y));
      lp.y = bf_pack2(v.z - bf_hi_f32(v.z), v.w - bf_hi_f32(v.w));
      *(uint2*)&lds_xh[row + 1][q * 4] = hp;
      *(uint2*)&lds_xl[row + 1][q * 4] = lp;
    }
    if (tid < 64) {                                // zero halo rows t=-1, t=128
      const int hr = (tid < 32) ? 0 : (XROWS - 1);
      const int i  = tid & 31;
      lds_xh[hr][i] = (__bf16)0.0f;
      lds_xl[hr][i] = (__bf16)0.0f;
    }
  }

  // ---- stage weights for group n directly into WMMA B-fragment layout ----
  // Thread pp handles fo = pp>>4, fi pair fi0=(pp&15)*2, all 3 taps:
  // 6 consecutive global floats, pure bit-op decode, paired b32 LDS stores.
  {
    const float* wbase = w + (size_t)n * (FOc * FIc * KK);  // 6144 floats
#pragma unroll
    for (int j = 0; j < 4; ++j) {
      const int pp  = tid + j * 256;        // 0..1023
      const int fo  = pp >> 4;              // 0..63
      const int fi0 = (pp & 15) << 1;       // 0,2,..,30
      const float* g = wbase + (size_t)pp * 6;   // (fo*32+fi0)*3
      float f[6];
#pragma unroll
      for (int i = 0; i < 6; ++i) f[i] = g[i];   // f[c] = (fi0,c), f[3+c] = (fi0+1,c)

      const int nt = fo >> 4;               // N-tile
      const int ln = (((fi0 >> 3) & 1) << 4) + (fo & 15);  // fragment lane
      const int e  = (fi0 & 7) + ((fi0 >> 4) << 3);  // even fragment element
#pragma unroll
      for (int c = 0; c < KK; ++c) {
        const float f0 = f[c], f1 = f[3 + c];
        const unsigned hp = bf_pack2(f0, f1);
        const unsigned lp = bf_pack2(f0 - bf_hi_f32(f0), f1 - bf_hi_f32(f1));
        __bf16* dst = &lds_b[(((c << 2) + nt) * 32 + ln) * BPITCH];
        *(unsigned*)(dst + e)      = hp;    // elements e, e+1 (hi)
        *(unsigned*)(dst + 16 + e) = lp;    // elements e, e+1 (lo)
      }
    }
  }
  __syncthreads();

  // ---- per-wave WMMA: 16 t-rows x 64 fo; hot loop = ds_load_b128 + wmma ----
  const int lane = tid & 31;
  const int wv   = tid >> 5;        // 0..7 -> t-tile
  const int hl   = lane >> 4;       // lane half (K-half selector)
  const int l16  = lane & 15;       // M (A/C row) or N (B/C col)
  const int t0   = wv * 16;

  v8f acc[4] = {v8f{}, v8f{}, v8f{}, v8f{}};

#pragma unroll
  for (int c = 0; c < KK; ++c) {
    // A tile, chunk c: lds_x row r = t0 + M + c; per-lane fragment = two 16B
    // segments at bf16 offsets hl*8 and 16+hl*8 (A 16x32 16-bit layout).
    const __bf16* ahp = &lds_xh[t0 + l16 + c][0];
    const __bf16* alp = &lds_xl[t0 + l16 + c][0];
    FragB AH, AL;
    AH.q[0] = *(const uint4*)(ahp + hl * 8);
    AH.q[1] = *(const uint4*)(ahp + 16 + hl * 8);
    AL.q[0] = *(const uint4*)(alp + hl * 8);
    AL.q[1] = *(const uint4*)(alp + 16 + hl * 8);

#pragma unroll
    for (int nt = 0; nt < 4; ++nt) {
      const __bf16* bp = &lds_b[(((c << 2) + nt) * 32 + lane) * BPITCH];
      FragB BH, BL;
      BH.q[0] = *(const uint4*)(bp + 0);
      BH.q[1] = *(const uint4*)(bp + 8);
      BL.q[0] = *(const uint4*)(bp + 16);
      BL.q[1] = *(const uint4*)(bp + 24);
      // 3-term split-precision accumulation: Ah*Bh + Ah*Bl + Al*Bh
      acc[nt] = __builtin_amdgcn_wmma_f32_16x16x32_bf16(
          false, AH.v, false, BH.v, (short)0, acc[nt], false, false);
      acc[nt] = __builtin_amdgcn_wmma_f32_16x16x32_bf16(
          false, AH.v, false, BL.v, (short)0, acc[nt], false, false);
      acc[nt] = __builtin_amdgcn_wmma_f32_16x16x32_bf16(
          false, AL.v, false, BH.v, (short)0, acc[nt], false, false);
    }
  }

  // ---- bias + store per C/D layout: VGPR j -> row M = j + 8*half, col = l16 ----
  float* obase = out + ((size_t)(b * Tt) * Nn + n) * FOc;
#pragma unroll
  for (int nt = 0; nt < 4; ++nt) {
    const int fo = nt * 16 + l16;
    const float bv = bias[n * FOc + fo];
#pragma unroll
    for (int j = 0; j < 8; ++j) {
      const int t = t0 + j + hl * 8;
      obase[(size_t)t * (Nn * FOc) + fo] = acc[nt][j] + bv;
    }
  }
}

extern "C" void kernel_launch(void* const* d_in, const int* in_sizes, int n_in,
                              void* d_out, int out_size, void* d_ws, size_t ws_size,
                              hipStream_t stream) {
  const float* x    = (const float*)d_in[0];
  const float* wt   = (const float*)d_in[1];
  const float* bias = (const float*)d_in[2];
  float* out        = (float*)d_out;
  dim3 grid(Bb, Nn);             // b fast so same-n blocks are adjacent in issue order
  multiconv1d_wmma_bf16x3<<<grid, 256, 0, stream>>>(x, wt, bias, out);
}
